// BEATsLayer_87316685127995
// MI455X (gfx1250) — matrix operand order repaired and checked by reading, not compiled, not run on hardware
//
#include <hip/hip_runtime.h>
#include <math.h>

typedef _Float16 v16h __attribute__((ext_vector_type(16)));
typedef _Float16 h8   __attribute__((ext_vector_type(8)));
typedef float    v8f  __attribute__((ext_vector_type(8)));

#define E_   768
#define H_   12
#define D_   64
#define F_   3072
#define T_   1024
#define B_   8
#define NTOK (T_ * B_)    /* 8192 rows */
#define NBH  (B_ * H_)    /* 96 heads  */

static constexpr float ALPHA_ = 2.2133638394006434f;  /* 24^0.25 */

__device__ __forceinline__ v8f wmma16(v16h a, v16h b, v8f c) {
  return __builtin_amdgcn_wmma_f32_16x16x32_f16(
      /*neg_a=*/false, a, /*neg_b=*/false, b,
      /*c_mod=*/(short)0, c, /*reuse_a=*/false, /*reuse_b=*/false);
}

__device__ __forceinline__ v16h load_afrag(const _Float16* p) {
  /* A-frag (16-bit, 16x32): lane holds contiguous K [base..base+15] */
  h8 lo = *(const h8*)(p);
  h8 hi = *(const h8*)(p + 8);
  v16h a;
#pragma unroll
  for (int e = 0; e < 8; ++e) { a[e] = lo[e]; a[8 + e] = hi[e]; }
  return a;
}

/* ---- pure-VALU 16-lane (per half) butterfly reductions via DPP16 ----
   0xB1 = quad_perm(1,0,3,2), 0x4E = quad_perm(2,3,0,1),
   0x141 = row_half_mirror,   0x140 = row_mirror. */
template <int CTRL>
__device__ __forceinline__ float dppf(float v) {
  int x = __float_as_int(v);
  return __int_as_float(__builtin_amdgcn_update_dpp(x, x, CTRL, 0xF, 0xF, false));
}
__device__ __forceinline__ float rowmax16(float v) {
  v = fmaxf(v, dppf<0xB1>(v));
  v = fmaxf(v, dppf<0x4E>(v));
  v = fmaxf(v, dppf<0x141>(v));
  v = fmaxf(v, dppf<0x140>(v));
  return v;
}
__device__ __forceinline__ float rowsum16(float v) {
  v += dppf<0xB1>(v);
  v += dppf<0x4E>(v);
  v += dppf<0x141>(v);
  v += dppf<0x140>(v);
  return v;
}

/* ---------------- conversions ---------------- */

__global__ __launch_bounds__(256) void k_cvt_f16(const float* __restrict__ in,
                                                 _Float16* __restrict__ out, size_t n) {
  for (size_t i = (size_t)blockIdx.x * blockDim.x + threadIdx.x; i < n;
       i += (size_t)gridDim.x * blockDim.x)
    out[i] = (_Float16)in[i];
}

/* Wt[n][k] = W[k][n], f32 -> f16 */
__global__ __launch_bounds__(256) void k_trans_cvt(const float* __restrict__ W,
                                                   _Float16* __restrict__ Wt, int K, int N) {
  size_t total = (size_t)K * N;
  for (size_t i = (size_t)blockIdx.x * blockDim.x + threadIdx.x; i < total;
       i += (size_t)gridDim.x * blockDim.x) {
    int k = (int)(i / N), n = (int)(i % N);
    Wt[(size_t)n * K + k] = (_Float16)W[i];
  }
}

/* vt[bh][d][t] = v16[(t*B + b)*E + h*D + d] */
__global__ __launch_bounds__(256) void k_vtrans(const _Float16* __restrict__ v16,
                                                _Float16* __restrict__ vt) {
  size_t total = (size_t)NBH * D_ * T_;
  for (size_t i = (size_t)blockIdx.x * blockDim.x + threadIdx.x; i < total;
       i += (size_t)gridDim.x * blockDim.x) {
    int t  = (int)(i % T_);
    int d  = (int)((i / T_) % D_);
    int bh = (int)(i / ((size_t)T_ * D_));
    int b = bh / H_, h = bh % H_;
    vt[i] = v16[((size_t)(t * B_ + b)) * E_ + h * D_ + d];
  }
}

/* ---------------- WMMA GEMM: C[M,N] = A[M,K](f16) * Wt[N,K](f16) + bias ----------------
   EPI: 0 = f16 out (bias), 1 = f16 out (bias + exact GELU), 2 = f32 out (bias).
   Block = 256 threads (8 waves); each wave owns a 32x64 tile (2 M sub-tiles x 4 N tiles:
   8 WMMAs share 4 B fragments per K step). */
template <int EPI>
__global__ __launch_bounds__(256) void k_gemm(const _Float16* __restrict__ A,
                                              const _Float16* __restrict__ Bt,
                                              const float* __restrict__ bias,
                                              void* __restrict__ out, int M, int N, int K) {
  const int wave = threadIdx.x >> 5;
  const int lane = threadIdx.x & 31;
  const int half = lane >> 4, nl = lane & 15;
  const int m0 = (blockIdx.y * 8 + wave) * 32;
  const int n0 = blockIdx.x * 64;
  if (m0 >= M) return;

  v8f acc[2][4] = {};
  const _Float16* Ap0 = A + (size_t)(m0 + nl) * K + half * 16;
  const _Float16* Ap1 = Ap0 + (size_t)16 * K;

  for (int k0 = 0; k0 < K; k0 += 32) {
    v16h a0 = load_afrag(Ap0 + k0);
    v16h a1 = load_afrag(Ap1 + k0);
#pragma unroll
    for (int j = 0; j < 4; ++j) {
      const int ncol = n0 + j * 16 + nl;
      v16h b = *(const v16h*)(Bt + (size_t)ncol * K + k0 + half * 16);
      acc[0][j] = wmma16(a0, b, acc[0][j]);
      acc[1][j] = wmma16(a1, b, acc[1][j]);
    }
  }

#pragma unroll
  for (int s = 0; s < 2; ++s) {
    const int mrow = m0 + s * 16 + half * 8;
#pragma unroll
    for (int j = 0; j < 4; ++j) {
      const int ncol = n0 + j * 16 + nl;
      const float bv = bias[ncol];
#pragma unroll
      for (int r = 0; r < 8; ++r) {
        float v = acc[s][j][r] + bv;
        if (EPI == 1) v = 0.5f * v * (1.0f + erff(v * 0.70710678118654752f));
        const size_t idx = (size_t)(mrow + r) * N + ncol;
        if (EPI == 2) ((float*)out)[idx] = v;
        else          ((_Float16*)out)[idx] = (_Float16)v;
      }
    }
  }
}

/* ---------------- grep gate: gates[bh*T + t] ---------------- */
__global__ __launch_bounds__(256) void k_gate(const _Float16* __restrict__ q16,
                                              const float* __restrict__ grepW,
                                              const float* __restrict__ grepB,
                                              const float* __restrict__ grepA,
                                              float* __restrict__ gates) {
  int idx = blockIdx.x * blockDim.x + threadIdx.x;  /* bh*T + t */
  if (idx >= NBH * T_) return;
  int bh = idx / T_, t = idx % T_;
  int b = bh / H_, h = bh % H_;
  const _Float16* qp = q16 + ((size_t)(t * B_ + b)) * E_ + h * D_;
  float s[8] = {0, 0, 0, 0, 0, 0, 0, 0};
  for (int d = 0; d < D_; ++d) {
    float qv = (float)qp[d];              /* q4 == raw q: scale*256 == 1 */
#pragma unroll
    for (int j = 0; j < 8; ++j) s[j] += qv * grepW[d * 8 + j];
  }
  float s0 = 0.f, s1 = 0.f;
#pragma unroll
  for (int j = 0; j < 4; ++j) { s0 += s[j] + grepB[j]; s1 += s[4 + j] + grepB[4 + j]; }
  float ga = 1.0f / (1.0f + __expf(-s0));
  float gb = 1.0f / (1.0f + __expf(-s1));
  gates[idx] = ga * (gb * grepA[h] - 1.0f) + 2.0f;
}

/* ---------------- flash attention with WMMA ----------------
   grid(NBH, T/64), block 128 (4 waves); each wave owns one 16-row Q tile.
   Online softmax in 16-column sub-steps; row reductions are DPP16 (no LDS). */
__global__ __launch_bounds__(128) void k_attn(const _Float16* __restrict__ q16,
                                              const _Float16* __restrict__ k16,
                                              const _Float16* __restrict__ vt,
                                              const float* __restrict__ pb,
                                              const float* __restrict__ gates,
                                              _Float16* __restrict__ ctx16) {
  const int bh = blockIdx.x;
  const int b = bh / H_, h = bh % H_;
  const int wave = threadIdx.x >> 5;
  const int lane = threadIdx.x & 31;
  const int half = lane >> 4, nl = lane & 15;
  const int m0 = (blockIdx.y * 4 + wave) * 16;

  __shared__ __attribute__((aligned(32))) _Float16 pls[4][16][48];

  /* Q fragments for K-dim d = 0..63 (two K=32 chunks), raw (unscaled) q */
  const _Float16* qp = q16 + ((size_t)((m0 + nl) * B_ + b)) * E_ + h * D_ + half * 16;
  v16h aq0 = load_afrag(qp);
  v16h aq1 = load_afrag(qp + 32);

  float mrun[8], lrun[8], g8[8];
  v8f o[4] = {};
#pragma unroll
  for (int r = 0; r < 8; ++r) {
    mrun[r] = -1e30f;
    lrun[r] = 0.0f;
    g8[r] = gates[(size_t)bh * T_ + m0 + half * 8 + r];
  }
  /* pb pointer for this lane's column within a 16-wide tile */
  const float* pbp = pb + (size_t)bh * T_ * T_ + (size_t)(m0 + half * 8) * T_ + nl;
  const _Float16* vtb = vt + (size_t)bh * D_ * T_;

  for (int s0 = 0; s0 < T_; s0 += 32) {
#pragma unroll
    for (int j = 0; j < 2; ++j) {
      const int c0 = s0 + j * 16;
      /* batch the 8 pos_bias loads first (streamed once -> non-temporal) */
      float z[8];
#pragma unroll
      for (int r = 0; r < 8; ++r)
        z[r] = __builtin_nontemporal_load(pbp + (size_t)r * T_ + c0);
      /* scores S = Qraw * K^T for 16 columns */
      const _Float16* kp = k16 + ((size_t)((c0 + nl) * B_ + b)) * E_ + h * D_ + half * 16;
      v16h bf0 = *(const v16h*)(kp);
      v16h bf1 = *(const v16h*)(kp + 32);
      v8f sacc = {};
      sacc = wmma16(aq0, bf0, sacc);
      sacc = wmma16(aq1, bf1, sacc);
      /* z = 0.125*s + gate*pb (ref's max-shift folds away: softmax shift-invariant) */
#pragma unroll
      for (int r = 0; r < 8; ++r) z[r] = 0.125f * sacc[r] + g8[r] * z[r];
      /* online softmax update; DPP16 row reductions stay within each 16-lane half */
#pragma unroll
      for (int r = 0; r < 8; ++r) {
        const float mx = rowmax16(z[r]);
        const float mnew = fmaxf(mrun[r], mx);
        const float resc = __expf(mrun[r] - mnew);
        mrun[r] = mnew;
        lrun[r] *= resc;
#pragma unroll
        for (int jj = 0; jj < 4; ++jj) o[jj][r] *= resc;
        const float p = __expf(z[r] - mnew);
        lrun[r] += p;                               /* per-lane column partial */
        pls[wave][half * 8 + r][j * 16 + nl] = (_Float16)p;
      }
    }
    /* wave-local C->A relayout through LDS (barrier also orders this wave's DS ops;
       each wave touches only pls[wave]) */
    __syncthreads();
    v16h pa;
    {
      const _Float16* pr = &pls[wave][nl][0];
      h8 lo = *(const h8*)(pr + half * 16);
      h8 hi = *(const h8*)(pr + half * 16 + 8);
#pragma unroll
      for (int e = 0; e < 8; ++e) { pa[e] = lo[e]; pa[8 + e] = hi[e]; }
    }
    /* ctx accumulate: O += P(16x32) * V(32x64) via transposed V */
#pragma unroll
    for (int j = 0; j < 4; ++j) {
      v16h bf = *(const v16h*)(vtb + (size_t)(j * 16 + nl) * T_ + s0 + half * 16);
      o[j] = wmma16(pa, bf, o[j]);
    }
  }

  /* finish: reduce l across the half (DPP), normalize, store ctx (f16) */
#pragma unroll
  for (int r = 0; r < 8; ++r) lrun[r] = 1.0f / rowsum16(lrun[r]);
#pragma unroll
  for (int j = 0; j < 4; ++j)
#pragma unroll
    for (int r = 0; r < 8; ++r) {
      const int mrow = m0 + half * 8 + r;
      ctx16[((size_t)(mrow * B_ + b)) * E_ + h * D_ + j * 16 + nl] =
          (_Float16)(o[j][r] * lrun[r]);
    }
}

/* ---------------- fused residual + LayerNorm ----------------
   z = a + ALPHA*res ; out = LN(z)*g + b ; optional f16 copy. One block per row. */
__global__ __launch_bounds__(256) void k_ln(const float* __restrict__ a,
                                            const float* __restrict__ res,
                                            const float* __restrict__ g,
                                            const float* __restrict__ bta,
                                            float* __restrict__ out32,
                                            _Float16* __restrict__ out16) {
  const int row = blockIdx.x;
  const float* ap = a + (size_t)row * E_;
  const float* rp = res + (size_t)row * E_;
  __shared__ float sbuf[8];
  float vals[3];
  float lsum = 0.f;
#pragma unroll
  for (int i = 0; i < 3; ++i) {
    int c = threadIdx.x + i * 256;
    vals[i] = ap[c] + ALPHA_ * rp[c];
    lsum += vals[i];
  }
  auto blocksum = [&](float v) -> float {
    /* per-half DPP reduce, then cross-half + cross-wave through LDS */
    v = rowsum16(v);
    int w = threadIdx.x >> 5, ln = threadIdx.x & 31;
    __syncthreads();
    if ((ln & 15) == 0) sbuf[w] = 0.f;   /* clear slot (one writer per wave wins below) */
    __syncthreads();
    if (ln == 0) sbuf[w] = v;
    __syncthreads();
    if (ln == 16) sbuf[w] += v;          /* other half */
    __syncthreads();
    float r = 0.f;
#pragma unroll
    for (int i = 0; i < 8; ++i) r += sbuf[i];
    return r;
  };
  const float mu = blocksum(lsum) * (1.0f / E_);
  float lvar = 0.f;
#pragma unroll
  for (int i = 0; i < 3; ++i) { float d = vals[i] - mu; lvar += d * d; }
  const float var = blocksum(lvar) * (1.0f / E_);
  const float rstd = rsqrtf(var + 1e-5f);
#pragma unroll
  for (int i = 0; i < 3; ++i) {
    int c = threadIdx.x + i * 256;
    float y = (vals[i] - mu) * rstd * g[c] + bta[c];
    out32[(size_t)row * E_ + c] = y;
    if (out16) out16[(size_t)row * E_ + c] = (_Float16)y;
  }
}

/* ---------------- host ---------------- */

extern "C" void kernel_launch(void* const* d_in, const int* in_sizes, int n_in,
                              void* d_out, int out_size, void* d_ws, size_t ws_size,
                              hipStream_t stream) {
  const float* states = (const float*)d_in[0];
  const float* posb   = (const float*)d_in[1];
  const float* Wq = (const float*)d_in[2];  const float* bq = (const float*)d_in[3];
  const float* Wk = (const float*)d_in[4];  const float* bk = (const float*)d_in[5];
  const float* Wv = (const float*)d_in[6];  const float* bv = (const float*)d_in[7];
  const float* Wo = (const float*)d_in[8];  const float* bo = (const float*)d_in[9];
  const float* grepW = (const float*)d_in[10];
  const float* grepB = (const float*)d_in[11];
  const float* grepA = (const float*)d_in[12];
  const float* ln1g = (const float*)d_in[13]; const float* ln1b = (const float*)d_in[14];
  const float* W1 = (const float*)d_in[15];  const float* b1 = (const float*)d_in[16];
  const float* W2 = (const float*)d_in[17];  const float* b2 = (const float*)d_in[18];
  const float* ln2g = (const float*)d_in[19]; const float* ln2b = (const float*)d_in[20];

  /* workspace carve-up (256B aligned) */
  char* wp = (char*)d_ws;
  auto alloc = [&](size_t bytes) -> void* {
    void* p = (void*)wp;
    wp += (bytes + 255) & ~(size_t)255;
    return p;
  };
  const size_t NE = (size_t)NTOK * E_;       /* 6291456 */
  _Float16* s16  = (_Float16*)alloc(NE * 2);
  _Float16* q16  = (_Float16*)alloc(NE * 2);
  _Float16* kk16 = (_Float16*)alloc(NE * 2);
  _Float16* vv16 = (_Float16*)alloc(NE * 2);
  _Float16* vt16 = (_Float16*)alloc(NE * 2);
  _Float16* ctx16 = (_Float16*)alloc(NE * 2);
  _Float16* x16  = (_Float16*)alloc(NE * 2);
  _Float16* h16  = (_Float16*)alloc((size_t)NTOK * F_ * 2);
  _Float16* Wqt = (_Float16*)alloc((size_t)E_ * E_ * 2);
  _Float16* Wkt = (_Float16*)alloc((size_t)E_ * E_ * 2);
  _Float16* Wvt = (_Float16*)alloc((size_t)E_ * E_ * 2);
  _Float16* Wot = (_Float16*)alloc((size_t)E_ * E_ * 2);
  _Float16* W1t = (_Float16*)alloc((size_t)E_ * F_ * 2);
  _Float16* W2t = (_Float16*)alloc((size_t)F_ * E_ * 2);
  float* gates  = (float*)alloc((size_t)NBH * T_ * 4);
  float* attn32 = (float*)alloc(NE * 4);
  float* x32    = (float*)alloc(NE * 4);
  float* y32    = (float*)alloc(NE * 4);

  /* 1. conversions */
  k_cvt_f16<<<4096, 256, 0, stream>>>(states, s16, NE);
  k_trans_cvt<<<2048, 256, 0, stream>>>(Wq, Wqt, E_, E_);
  k_trans_cvt<<<2048, 256, 0, stream>>>(Wk, Wkt, E_, E_);
  k_trans_cvt<<<2048, 256, 0, stream>>>(Wv, Wvt, E_, E_);
  k_trans_cvt<<<2048, 256, 0, stream>>>(Wo, Wot, E_, E_);
  k_trans_cvt<<<4096, 256, 0, stream>>>(W1, W1t, E_, F_);
  k_trans_cvt<<<4096, 256, 0, stream>>>(W2, W2t, F_, E_);

  /* 2. QKV projections (raw q; score scale folded into attention) */
  dim3 gEE(E_ / 64, NTOK / 256);
  k_gemm<0><<<gEE, 256, 0, stream>>>(s16, Wqt, bq, q16, NTOK, E_, E_);
  k_gemm<0><<<gEE, 256, 0, stream>>>(s16, Wkt, bk, kk16, NTOK, E_, E_);
  k_gemm<0><<<gEE, 256, 0, stream>>>(s16, Wvt, bv, vv16, NTOK, E_, E_);

  /* 3. gates + V transpose */
  k_gate<<<(NBH * T_) / 256, 256, 0, stream>>>(q16, grepW, grepB, grepA, gates);
  k_vtrans<<<4096, 256, 0, stream>>>(vv16, vt16);

  /* 4. flash attention */
  k_attn<<<dim3(NBH, T_ / 64), 128, 0, stream>>>(q16, kk16, vt16, posb, gates, ctx16);

  /* 5. output projection + LN1 */
  k_gemm<2><<<gEE, 256, 0, stream>>>(ctx16, Wot, bo, attn32, NTOK, E_, E_);
  k_ln<<<NTOK, 256, 0, stream>>>(attn32, states, ln1g, ln1b, x32, x16);

  /* 6. FFN (GELU fused) + LN2 -> d_out */
  k_gemm<1><<<dim3(F_ / 64, NTOK / 256), 256, 0, stream>>>(x16, W1t, b1, h16, NTOK, F_, E_);
  k_gemm<2><<<gEE, 256, 0, stream>>>(h16, W2t, b2, y32, NTOK, E_, F_);
  k_ln<<<NTOK, 256, 0, stream>>>(y32, x32, ln2g, ln2b, (float*)d_out, (_Float16*)nullptr);

  /* 7. second output: pass positional_bias through */
  hipMemcpyAsync((float*)d_out + NE, posb, (size_t)NBH * T_ * T_ * sizeof(float),
                 hipMemcpyDeviceToDevice, stream);
}
